// Block_42339787604393
// MI455X (gfx1250) — matrix operand order repaired and checked by reading, not compiled, not run on hardware
//
#include <hip/hip_runtime.h>
#include <hip/hip_bf16.h>
#include <math.h>

typedef __attribute__((ext_vector_type(16))) _Float16 v16h;
typedef __attribute__((ext_vector_type(8)))  float    v8f;
typedef unsigned int  u32;
typedef unsigned short u16;

union FragU { v16h h; u32 u[8]; };

static __device__ __forceinline__ v8f wmma_f16(v16h a, v16h b, v8f c) {
  // D = A(16x32 f16) x B(32x16 f16) + C(16x16 f32)
  return __builtin_amdgcn_wmma_f32_16x16x32_f16(false, a, false, b, (short)0, c,
                                                false, false);
}

// -------------------------------------------------------------------------
// Weight convert: f32 row-major [K x N] -> f16 "paired-K" layout:
// u32[kk * N + n] = { w[2kk][n] (lo), w[2kk+1][n] (hi) }  -- exactly the
// B-fragment pairing for v_wmma_f32_16x16x32_f16.
// -------------------------------------------------------------------------
__global__ void cvt_pack_kernel(const float* __restrict__ w, u32* __restrict__ dst,
                                int K, int N) {
  int i = blockIdx.x * blockDim.x + threadIdx.x;
  int total = (K >> 1) * N;
  if (i >= total) return;
  int kk = i / N, n = i - kk * N;
  union { u32 u; _Float16 h[2]; } p;
  p.h[0] = (_Float16)w[(2 * kk) * N + n];
  p.h[1] = (_Float16)w[(2 * kk + 1) * N + n];
  dst[i] = p.u;
}

// -------------------------------------------------------------------------
// LayerNorm: one row (768) per workgroup, f32 in -> f16 out
// -------------------------------------------------------------------------
__global__ __launch_bounds__(256) void ln_kernel(const float* __restrict__ x,
                                                 const float* __restrict__ g,
                                                 const float* __restrict__ b,
                                                 _Float16* __restrict__ out) {
  __shared__ float rs[256], rq[256];
  int row = blockIdx.x, tid = threadIdx.x;
  const float* xr = x + (size_t)row * 768;
  float s = 0.f, q = 0.f;
  for (int i = tid; i < 768; i += 256) { float v = xr[i]; s += v; q += v * v; }
  rs[tid] = s; rq[tid] = q;
  __syncthreads();
  for (int off = 128; off > 0; off >>= 1) {
    if (tid < off) { rs[tid] += rs[tid + off]; rq[tid] += rq[tid + off]; }
    __syncthreads();
  }
  float mu  = rs[0] * (1.f / 768.f);
  float var = rq[0] * (1.f / 768.f) - mu * mu;
  float inv = rsqrtf(var + 1e-5f);
  _Float16* orow = out + (size_t)row * 768;
  for (int i = tid; i < 768; i += 256)
    orow[i] = (_Float16)((xr[i] - mu) * inv * g[i] + b[i]);
}

// -------------------------------------------------------------------------
// WMMA GEMM: C[M,N] = A[M,K](f16 rowmajor) * B[K,N](paired-f16) (+bias)
//            (optional exact GELU) (+optional f32 residual), store f32 or f16
// Tile 128x128, Kstep 32, 256 threads = 8 waves (4 M-waves x 2 N-waves),
// each wave owns a 32x64 patch = 2x4 WMMA accumulators.
// b128 staging, register double-buffer, global_prefetch for K+2 tile.
// LDS strides: A 20 u32/row (80B, 16B-aligned quads, conflict-free frag reads)
//              B 132 u32/row (528B)
// -------------------------------------------------------------------------
__global__ __launch_bounds__(256) void gemm_kernel(
    const _Float16* __restrict__ A, const u32* __restrict__ Bp,
    const float* __restrict__ bias, const float* __restrict__ resid,
    float* __restrict__ outF, _Float16* __restrict__ outH,
    int M, int N, int K, int do_gelu) {
  __shared__ __align__(16) u32 lA[128 * 20];   // 10,240 B
  __shared__ __align__(16) u32 lB[16 * 132];   //  8,448 B
  const u32* Au = (const u32*)A;
  int tid = threadIdx.x, lane = tid & 31;
  int wave = tid >> 5, half = lane >> 4, l15 = lane & 15;
  int wm = wave >> 1, wn = wave & 1;
  int m0 = blockIdx.y * 128, n0 = blockIdx.x * 128;

  // staging coordinates (each thread moves 2x uint4 for A and 2x for B)
  int arow = tid >> 2, aquad = tid & 3;     // A rows arow, arow+64 ; 4 quads/row
  int brow = tid >> 5, bquad = tid & 31;    // B rows brow, brow+8  ; 32 quads/row
  const int KH = K >> 1;                    // u32 per A row

  v8f acc[2][4];
  v8f z = {};
  for (int s = 0; s < 2; ++s)
    for (int t = 0; t < 4; ++t) acc[s][t] = z;

  const int nk = K >> 5;
  // prologue: fetch tile 0 into registers
  uint4 pa0, pa1, pb0, pb1;
  {
    size_t ea = (size_t)(m0 + arow) * KH + aquad * 4;
    pa0 = *(const uint4*)(Au + ea);
    pa1 = *(const uint4*)(Au + ea + (size_t)64 * KH);
    size_t eb = (size_t)brow * N + n0 + bquad * 4;
    pb0 = *(const uint4*)(Bp + eb);
    pb1 = *(const uint4*)(Bp + eb + (size_t)8 * N);
  }

  for (int kt = 0; kt < nk; ++kt) {
    // commit current tile registers to LDS
    *(uint4*)&lA[arow * 20 + aquad * 4] = pa0;
    *(uint4*)&lA[(arow + 64) * 20 + aquad * 4] = pa1;
    *(uint4*)&lB[brow * 132 + bquad * 4] = pb0;
    *(uint4*)&lB[(brow + 8) * 132 + bquad * 4] = pb1;
    __syncthreads();

    // kick off next tile's global loads (overlap with WMMA below)
    if (kt + 1 < nk) {
      size_t ea = (size_t)(m0 + arow) * KH + ((kt + 1) << 4) + aquad * 4;
      pa0 = *(const uint4*)(Au + ea);
      pa1 = *(const uint4*)(Au + ea + (size_t)64 * KH);
      size_t eb = (size_t)((kt + 1) * 16 + brow) * N + n0 + bquad * 4;
      pb0 = *(const uint4*)(Bp + eb);
      pb1 = *(const uint4*)(Bp + eb + (size_t)8 * N);
    }
    if (kt + 2 < nk) {  // keep L2/WGP$ ahead -> global_prefetch_b8
      __builtin_prefetch(Au + (size_t)(m0 + arow) * KH + ((kt + 2) << 4) + aquad * 4, 0, 1);
      __builtin_prefetch(Bp + (size_t)((kt + 2) * 16 + brow) * N + n0 + bquad * 4, 0, 1);
    }

    v16h af[2], bf[4];
#pragma unroll
    for (int s = 0; s < 2; ++s) {            // A frags (doc: grouped K layout)
      FragU f;
      int m = wm * 32 + s * 16 + l15;
#pragma unroll
      for (int v = 0; v < 8; ++v) {
        int kk = (v >> 2) * 8 + half * 4 + (v & 3);
        f.u[v] = lA[m * 20 + kk];
      }
      af[s] = f.h;
    }
#pragma unroll
    for (int t = 0; t < 4; ++t) {            // B frags (lane=col, half=K half)
      FragU f;
      int n = wn * 64 + t * 16 + l15;
#pragma unroll
      for (int v = 0; v < 8; ++v) {
        int kk = (v >> 2) * 8 + half * 4 + (v & 3);
        f.u[v] = lB[kk * 132 + n];
      }
      bf[t] = f.h;
    }
#pragma unroll
    for (int s = 0; s < 2; ++s)
#pragma unroll
      for (int t = 0; t < 4; ++t)
        acc[s][t] = wmma_f16(af[s], bf[t], acc[s][t]);
    __syncthreads();
  }

  // epilogue: C/D layout lane=col(l15), row = r + 8*half
#pragma unroll
  for (int s = 0; s < 2; ++s) {
#pragma unroll
    for (int t = 0; t < 4; ++t) {
      int col = n0 + wn * 64 + t * 16 + l15;
      float bcol = bias ? bias[col] : 0.f;
      int rowb = m0 + wm * 32 + s * 16 + half * 8;
      float vals[8];
#pragma unroll
      for (int r = 0; r < 8; ++r) {
        float v = acc[s][t][r] + bcol;
        if (do_gelu) v = 0.5f * v * (1.0f + erff(v * 0.70710678118654752f));
        vals[r] = v;
      }
      if (resid) {
#pragma unroll
        for (int r = 0; r < 8; ++r)
          vals[r] += resid[(size_t)(rowb + r) * N + col];
      }
      if (outF) {
#pragma unroll
        for (int r = 0; r < 8; ++r)
          outF[(size_t)(rowb + r) * N + col] = vals[r];
      } else {
#pragma unroll
        for (int r = 0; r < 8; ++r)
          outH[(size_t)(rowb + r) * N + col] = (_Float16)vals[r];
      }
    }
  }
}

// -------------------------------------------------------------------------
// Attention (two-pass, WMMA): one workgroup = (batch, head, 16 query rows),
// 4 waves split the 1024 keys. qkv layout: (B,N,3,H,64) f16.
// -------------------------------------------------------------------------
static __device__ __forceinline__ v16h load_k_bfrag(const u32* qu, int bb, int hd,
                                                    int n, int f, int half) {
  // B[k=d][col=key]: (d,d+1) pairs are contiguous in memory for fixed key
  size_t base = (size_t)(bb * 1024 + n) * 2304 + 768 + hd * 64;
  FragU fr;
#pragma unroll
  for (int v = 0; v < 8; ++v) {
    int d = f * 32 + (v >> 2) * 16 + half * 8 + (v & 3) * 2;
    fr.u[v] = qu[(base + d) >> 1];
  }
  return fr.h;
}

__global__ __launch_bounds__(128) void attn_kernel(const _Float16* __restrict__ qkv,
                                                   _Float16* __restrict__ out) {
  __shared__ float maxw[4][16];
  __shared__ float rowmax[16];
  __shared__ float lw[4][16];
  __shared__ _Float16 plds[4 * 16 * 34];     // per-wave P tile 16x32 (+pad)
  __shared__ float opart[4][16][64];         // per-wave partial O
  const u32* qu = (const u32*)qkv;
  const u16* qs = (const u16*)qkv;
  int tid = threadIdx.x, lane = tid & 31;
  int wave = tid >> 5, half = lane >> 4, l15 = lane & 15;
  int q0 = blockIdx.x * 16;
  int hd = blockIdx.y;
  int bb = blockIdx.z;
  const float scale = 0.125f;                // 64^-0.5

  // Q A-fragments (rows=queries, K=head dim, split d:0-31 / 32-63)
  v16h aq[2];
  {
    size_t base = (size_t)(bb * 1024 + q0 + l15) * 2304 + hd * 64;
#pragma unroll
    for (int f = 0; f < 2; ++f) {
      FragU fr;
#pragma unroll
      for (int v = 0; v < 8; ++v) {
        int d = f * 32 + (v >> 2) * 16 + half * 8 + (v & 3) * 2;
        fr.u[v] = qu[(base + d) >> 1];
      }
      aq[f] = fr.h;
    }
  }

  // ---- pass A: row maxima of scaled scores ----
  float rmax[8];
#pragma unroll
  for (int r = 0; r < 8; ++r) rmax[r] = -1e30f;
  for (int kt = wave * 16; kt < 1024; kt += 64) {
    v16h bk0 = load_k_bfrag(qu, bb, hd, kt + l15, 0, half);
    v16h bk1 = load_k_bfrag(qu, bb, hd, kt + l15, 1, half);
    v8f zz = {};
    v8f sacc = wmma_f16(aq[0], bk0, zz);
    sacc = wmma_f16(aq[1], bk1, sacc);
#pragma unroll
    for (int r = 0; r < 8; ++r) rmax[r] = fmaxf(rmax[r], sacc[r] * scale);
  }
  for (int off = 1; off < 16; off <<= 1)
#pragma unroll
    for (int r = 0; r < 8; ++r)
      rmax[r] = fmaxf(rmax[r], __shfl_xor(rmax[r], off, 32));
  if (l15 == 0) {
#pragma unroll
    for (int r = 0; r < 8; ++r) maxw[wave][half * 8 + r] = rmax[r];
  }
  __syncthreads();
  if (tid < 16) {
    float m = fmaxf(fmaxf(maxw[0][tid], maxw[1][tid]),
                    fmaxf(maxw[2][tid], maxw[3][tid]));
    rowmax[tid] = m;
  }
  __syncthreads();
  float rm[8];
#pragma unroll
  for (int r = 0; r < 8; ++r) rm[r] = rowmax[half * 8 + r];

  // ---- pass B: P = exp(S - max); O += P.V; track row sums ----
  v8f oacc[4];
  v8f z2 = {};
#pragma unroll
  for (int dt = 0; dt < 4; ++dt) oacc[dt] = z2;
  float lsum[8];
#pragma unroll
  for (int r = 0; r < 8; ++r) lsum[r] = 0.f;
  u32* pu = (u32*)plds;

  for (int kt = wave * 32; kt < 1024; kt += 128) {
#pragma unroll
    for (int t = 0; t < 2; ++t) {
      int kc = kt + t * 16;
      v16h bk0 = load_k_bfrag(qu, bb, hd, kc + l15, 0, half);
      v16h bk1 = load_k_bfrag(qu, bb, hd, kc + l15, 1, half);
      v8f zz = {};
      v8f sacc = wmma_f16(aq[0], bk0, zz);
      sacc = wmma_f16(aq[1], bk1, sacc);
#pragma unroll
      for (int r = 0; r < 8; ++r) {
        float p = __expf(sacc[r] * scale - rm[r]);
        lsum[r] += p;
        plds[wave * 544 + (r + half * 8) * 34 + t * 16 + l15] = (_Float16)p;
      }
    }
    // reshape P (C layout) -> A fragment via per-wave LDS bounce
    v16h pf;
    {
      FragU fr;
#pragma unroll
      for (int v = 0; v < 8; ++v) {
        int kk = (v >> 2) * 8 + half * 4 + (v & 3);
        fr.u[v] = pu[wave * 272 + l15 * 17 + kk];
      }
      pf = fr.h;
    }
    // V B-fragments: k-dim = keys (pairs along key -> two u16 gathers)
#pragma unroll
    for (int dt = 0; dt < 4; ++dt) {
      FragU fr;
      int d = dt * 16 + l15;
#pragma unroll
      for (int v = 0; v < 8; ++v) {
        int key = kt + (v >> 2) * 16 + half * 8 + (v & 3) * 2;
        size_t e0 = (size_t)(bb * 1024 + key) * 2304 + 1536 + hd * 64 + d;
        fr.u[v] = (u32)qs[e0] | ((u32)qs[e0 + 2304] << 16);
      }
      oacc[dt] = wmma_f16(pf, fr.h, oacc[dt]);
    }
  }

  for (int off = 1; off < 16; off <<= 1)
#pragma unroll
    for (int r = 0; r < 8; ++r) lsum[r] += __shfl_xor(lsum[r], off, 32);
  if (l15 == 0) {
#pragma unroll
    for (int r = 0; r < 8; ++r) lw[wave][half * 8 + r] = lsum[r];
  }
#pragma unroll
  for (int dt = 0; dt < 4; ++dt)
#pragma unroll
    for (int r = 0; r < 8; ++r)
      opart[wave][r + half * 8][dt * 16 + l15] = oacc[dt][r];
  __syncthreads();

  for (int i = tid; i < 1024; i += 128) {
    int m = i >> 6, d = i & 63;
    float v = opart[0][m][d] + opart[1][m][d] + opart[2][m][d] + opart[3][m][d];
    float l = lw[0][m] + lw[1][m] + lw[2][m] + lw[3][m];
    out[(size_t)(bb * 1024 + q0 + m) * 768 + hd * 64 + d] = (_Float16)(v / l);
  }
}

// -------------------------------------------------------------------------
// Host orchestration
// -------------------------------------------------------------------------
extern "C" void kernel_launch(void* const* d_in, const int* in_sizes, int n_in,
                              void* d_out, int out_size, void* d_ws, size_t ws_size,
                              hipStream_t stream) {
  const float* x      = (const float*)d_in[0];
  const float* g1     = (const float*)d_in[1];
  const float* b1     = (const float*)d_in[2];
  const float* g2     = (const float*)d_in[3];
  const float* b2     = (const float*)d_in[4];
  const float* w_qkv  = (const float*)d_in[5];
  const float* w_proj = (const float*)d_in[6];
  const float* b_proj = (const float*)d_in[7];
  const float* w_fc1  = (const float*)d_in[8];
  const float* b_fc1  = (const float*)d_in[9];
  const float* w_fc2  = (const float*)d_in[10];
  const float* b_fc2  = (const float*)d_in[11];

  char* ws = (char*)d_ws;
  _Float16* h16    = (_Float16*)(ws + 0);                   // 12,582,912 B (reused for LN2)
  _Float16* qkv16  = (_Float16*)(ws + 12582912);            // 37,748,736 B
  _Float16* attn16 = (_Float16*)(ws + 12582912 + 37748736); // 12,582,912 B
  _Float16* g16    = (_Float16*)(ws + 12582912);            // reuses qkv16+attn16 (50,331,648 B)
  float*    x1     = (float*)(ws + 62914560);               // 25,165,824 B
  u32* wqkvp = (u32*)(ws + 88080384);                       //  3,538,944 B
  u32* wprojp= (u32*)(ws + 91619328);                       //  1,179,648 B
  u32* wfc1p = (u32*)(ws + 92798976);                       //  4,718,592 B
  u32* wfc2p = (u32*)(ws + 97517568);                       //  4,718,592 B (end ~102.2 MB)
  float* outF = (float*)d_out;

  cvt_pack_kernel<<<(384 * 2304 + 255) / 256, 256, 0, stream>>>(w_qkv, wqkvp, 768, 2304);
  cvt_pack_kernel<<<(384 * 768  + 255) / 256, 256, 0, stream>>>(w_proj, wprojp, 768, 768);
  cvt_pack_kernel<<<(384 * 3072 + 255) / 256, 256, 0, stream>>>(w_fc1, wfc1p, 768, 3072);
  cvt_pack_kernel<<<(1536 * 768 + 255) / 256, 256, 0, stream>>>(w_fc2, wfc2p, 3072, 768);

  ln_kernel<<<8192, 256, 0, stream>>>(x, g1, b1, h16);
  gemm_kernel<<<dim3(18, 64), 256, 0, stream>>>(h16, wqkvp, nullptr, nullptr,
                                                nullptr, qkv16, 8192, 2304, 768, 0);
  attn_kernel<<<dim3(64, 12, 8), 128, 0, stream>>>(qkv16, attn16);
  gemm_kernel<<<dim3(6, 64), 256, 0, stream>>>(attn16, wprojp, b_proj, x,
                                               x1, nullptr, 8192, 768, 768, 0);
  ln_kernel<<<8192, 256, 0, stream>>>(x1, g2, b2, h16);
  gemm_kernel<<<dim3(24, 64), 256, 0, stream>>>(h16, wfc1p, b_fc1, nullptr,
                                                nullptr, g16, 8192, 3072, 768, 1);
  gemm_kernel<<<dim3(6, 64), 256, 0, stream>>>(g16, wfc2p, b_fc2, x1,
                                               outF, nullptr, 8192, 768, 3072, 0);
}
